// GaussianKDE_10831907520620
// MI455X (gfx1250) — compile-verified, matrix operand order checked by reading, not run on hardware
//
#include <hip/hip_runtime.h>

// GaussianKDE on gfx1250 (MI455X, wave32).
// Masked reduction over pixels on the matrix unit via V_WMMA_F32_16X16X32_F16
// with compensated (hi+residual) f16 operands accumulated into two
// independent f32 accumulators. Masks are converted to f16 once at staging
// and fed to WMMA B via 16-byte LDS loads. (The 2x-unrolled variant was
// rejected: register pressure caused duplicated f16 converts per element.)

typedef __attribute__((ext_vector_type(16))) _Float16 v16h;
typedef __attribute__((ext_vector_type(8)))  _Float16 v8h;
typedef __attribute__((ext_vector_type(8)))  float    v8f;

#define NBIN     128
#define P_TOT    50176          // 224*224
#define CHUNK    512            // pixels staged in LDS per block
#define SLICES   98             // P_TOT / CHUNK
#define BC_TOT   24             // B*C = 8*3
#define CONST1F  0.19947114020071635f    // (2*pi*4)^-0.5, folded into normalize
#define NEGK     (-0.18033688011112042f) // -log2(e)/CONST2 = -log2(e)/8

// ws layout (floats): [0 .. 24*128) bin accumulators, [24*128 .. +8) mask sums
#define WS_MSUM  (BC_TOT * NBIN)

__global__ __launch_bounds__(256) void kde_partial_kernel(
    const float* __restrict__ images,   // [B*C, P]
    const float* __restrict__ masks,    // [B, P]
    const float* __restrict__ colors,   // [128]
    float* __restrict__ ws)
{
    __shared__ float lx[CHUNK];                       // x values, f32
    __shared__ __align__(16) _Float16 lmh[CHUNK];     // masks pre-converted to f16
    __shared__ float msum_sh;

    const int tid   = threadIdx.x;
    const int bid   = blockIdx.x;
    const int bc    = bid / SLICES;     // 0..23  (b*3 + c)
    const int slice = bid % SLICES;     // 0..97
    const int b_idx = bc / 3;
    const int c_idx = bc % 3;

    const float* Xp = images + (size_t)bc    * P_TOT + (size_t)slice * CHUNK;
    const float* Mp = masks  + (size_t)b_idx * P_TOT + (size_t)slice * CHUNK;

    if (tid == 0) msum_sh = 0.0f;
    __syncthreads();

    // ---- stage 512 pixels: x (f32) and mask (converted once to f16) ----
    float mloc = 0.0f;
    for (int i = tid; i < CHUNK; i += 256) {
        float xv = Xp[i];
        float mv = Mp[i];
        lx[i]  = xv;
        lmh[i] = (_Float16)mv;   // 0/1 exact in f16
        mloc  += mv;
    }
    if (c_idx == 0) atomicAdd(&msum_sh, mloc);   // ds_add_f32
    __syncthreads();

    // ---- per-wave: 16 bins, iterate 16 sub-chunks of 32 pixels ----
    const int lane    = tid & 31;
    const int wave    = tid >> 5;        // 0..7 -> bin group
    const int m       = lane & 15;       // WMMA A row (bin within group)
    const int hi      = lane >> 4;       // half-wave K offset selector
    const int binBase = wave * 16;
    const float col   = colors[binBase + m];

    v8f acc_h = {};   // accumulates f16-rounded kernel values
    v8f acc_l = {};   // accumulates residuals (compensation)
    for (int sub = 0; sub < CHUNK / 32; ++sub) {
        const int base = sub * 32 + 8 * hi;

        // B matrix: 16 mask halves for this lane's K indices, two b128 loads.
        v8h b0 = *(const v8h*)&lmh[base];        // K = base .. base+7
        v8h b1 = *(const v8h*)&lmh[base + 16];   // K = base+16 .. base+23
        v16h bmat;
        #pragma unroll
        for (int i = 0; i < 8; ++i) { bmat[i] = b0[i]; bmat[8 + i] = b1[i]; }

        v16h a_hi, a_lo;
        #pragma unroll
        for (int j = 0; j < 8; ++j) {
            // ISA 16-bit A 16x32 layout: VGPR j<4 -> K=2j(+8*hi),
            // VGPR j>=4 -> K=16+2(j-4)(+8*hi); two packed K per VGPR.
            const int p0 = base + ((j < 4) ? 0 : 16) + (j & 3) * 2;
            float x0 = lx[p0], x1 = lx[p0 + 1];       // LDS broadcast reads
            float d0 = x0 - col, d1 = x1 - col;
            float e0 = __builtin_amdgcn_exp2f(d0 * d0 * NEGK);  // v_exp_f32
            float e1 = __builtin_amdgcn_exp2f(d1 * d1 * NEGK);
            _Float16 h0 = (_Float16)e0, h1 = (_Float16)e1;
            a_hi[2*j]   = h0;
            a_hi[2*j+1] = h1;
            a_lo[2*j]   = (_Float16)(e0 - (float)h0);  // residual (v_fma_mix)
            a_lo[2*j+1] = (_Float16)(e1 - (float)h1);
        }
        // Independent accumulators -> no D->C hazard between the two WMMAs.
        acc_h = __builtin_amdgcn_wmma_f32_16x16x32_f16(
                    false, a_hi, false, bmat, (short)0, acc_h, false, false);
        acc_l = __builtin_amdgcn_wmma_f32_16x16x32_f16(
                    false, a_lo, false, bmat, (short)0, acc_l, false, false);
    }

    v8f acc = acc_h + acc_l;

    // D f32 layout: lane n==0 (lanes 0,16) hold rows binBase + 8*hi + j in VGPR j
    if ((lane & 15) == 0) {
        #pragma unroll
        for (int j = 0; j < 8; ++j) {
            int bin = binBase + hi * 8 + j;
            atomicAdd(&ws[bc * NBIN + bin], acc[j]);
        }
    }
    if (c_idx == 0 && tid == 0) {
        atomicAdd(&ws[WS_MSUM + b_idx], msum_sh);
    }
}

__global__ void kde_norm_kernel(const float* __restrict__ ws,
                                float* __restrict__ out)
{
    int i = blockIdx.x * blockDim.x + threadIdx.x;
    if (i < BC_TOT * NBIN) {
        int b = i / (3 * NBIN);
        float msum = ws[WS_MSUM + b];
        float v = ws[i] * CONST1F;           // CONST1 folded in here
        out[i] = (msum != 0.0f) ? (v / msum) : v;
    }
}

extern "C" void kernel_launch(void* const* d_in, const int* in_sizes, int n_in,
                              void* d_out, int out_size, void* d_ws, size_t ws_size,
                              hipStream_t stream) {
    const float* images = (const float*)d_in[0];  // [8,3,224,224] f32
    const float* masks  = (const float*)d_in[1];  // [8,1,224,224] f32
    const float* colors = (const float*)d_in[2];  // [128] f32
    float* ws  = (float*)d_ws;
    float* out = (float*)d_out;                   // [8,3,128] f32

    // zero accumulators each call (graph-capturable memset node)
    hipMemsetAsync(ws, 0, (size_t)(WS_MSUM + 8) * sizeof(float), stream);

    kde_partial_kernel<<<BC_TOT * SLICES, 256, 0, stream>>>(images, masks, colors, ws);
    kde_norm_kernel<<<(BC_TOT * NBIN + 255) / 256, 256, 0, stream>>>(ws, out);
}
